// AdmittanceGNN_66228395704524
// MI455X (gfx1250) — matrix operand order repaired and checked by reading, not compiled, not run on hardware
//
#include <hip/hip_runtime.h>

#define NN   50000
#define EE   640000
#define DD   128
#define LL   3
#define AH   64   // ATT_H

typedef __attribute__((ext_vector_type(16))) _Float16 v16h;
typedef __attribute__((ext_vector_type(8)))  _Float16 v8h;
typedef __attribute__((ext_vector_type(8)))  float    v8f;

#define WMMA_F32_F16(A, B, C) \
  __builtin_amdgcn_wmma_f32_16x16x32_f16(false, (A), false, (B), (short)0, (C), false, false)

// A-operand (16x32 f16, M x K): lane L holds row M = L&15; K-halves chosen by L>>4.
// VGPR0..7 = K(s+16*hi)..+15 contiguous (2 halves per VGPR).
static __device__ __forceinline__ v16h load_a_tile(const _Float16* rowp, int s, int hi) {
  v8h a = *(const v8h*)(rowp + s + 16 * hi);
  v8h b = *(const v8h*)(rowp + s + 16 * hi + 8);
  v16h r;
#pragma unroll
  for (int j = 0; j < 8; ++j) { r[j] = a[j]; r[j + 8] = b[j]; }
  return r;
}

// B-operand (32x16 f16, K x N): lane L holds column n = L&15, 16 consecutive K.
// p points at wt[col*128 + s + 16*hi]; works for LDS pointers (ds_load_b128 after
// address-space inference) and global pointers alike.
static __device__ __forceinline__ v16h load_b_16k(const _Float16* p) {
  v8h a = *(const v8h*)(p);
  v8h b = *(const v8h*)(p + 8);
  v16h r;
#pragma unroll
  for (int j = 0; j < 8; ++j) { r[j] = a[j]; r[j + 8] = b[j]; }
  return r;
}

// ---------------- prep kernels ----------------

__global__ void __launch_bounds__(256) prep_weights_kernel(
    const float* __restrict__ wn,   // [L,128,128]
    const float* __restrict__ we,   // [L,130,128]
    const float* __restrict__ w1,   // [L,256,64]
    _Float16* __restrict__ wnT,     // [L,128(out),128(K)]
    _Float16* __restrict__ weT,     // [L,128(out),128(K)]
    float* __restrict__ we2f,       // [L,2,128]
    _Float16* __restrict__ w1aT,    // [L,64(out),128(K)]
    _Float16* __restrict__ w1bT)    // [L,64(out),128(K)]
{
  int idx = blockIdx.x * blockDim.x + threadIdx.x;
  if (idx >= LL * DD * DD) return;
  int i = idx / (DD * DD);
  int r = idx - i * DD * DD;
  int k = r / DD;
  int n = r - k * DD;
  wnT[(size_t)i * DD * DD + (size_t)n * DD + k] = (_Float16)wn[(size_t)i * DD * DD + (size_t)k * DD + n];
  weT[(size_t)i * DD * DD + (size_t)n * DD + k] = (_Float16)we[(size_t)i * 130 * DD + (size_t)k * DD + n];
  if (k < 2)
    we2f[i * 2 * DD + k * DD + n] = we[(size_t)i * 130 * DD + (size_t)(DD + k) * DD + n];
  if (n < AH) {
    w1aT[(size_t)i * AH * DD + (size_t)n * DD + k] = (_Float16)w1[(size_t)i * 2 * DD * AH + (size_t)k * AH + n];
    w1bT[(size_t)i * AH * DD + (size_t)n * DD + k] = (_Float16)w1[(size_t)i * 2 * DD * AH + (size_t)(DD + k) * AH + n];
  }
}

__global__ void __launch_bounds__(256) f32_to_f16_kernel(const float* __restrict__ in,
                                                         _Float16* __restrict__ out, int n) {
  int i = blockIdx.x * blockDim.x + threadIdx.x;
  if (i < n) out[i] = (_Float16)in[i];
}

__global__ void __launch_bounds__(256) zero_kernel(float4* __restrict__ p, int n4) {
  int i = blockIdx.x * blockDim.x + threadIdx.x;
  if (i < n4) p[i] = make_float4(0.f, 0.f, 0.f, 0.f);
}

// ---------------- node linear: h16 = f16(x16 @ Wn) ----------------
// One wave computes a full 16x128 row block (8 output tiles); Wn staged in LDS.

__global__ void __launch_bounds__(256) node_linear_kernel(
    const _Float16* __restrict__ x16, const _Float16* __restrict__ wnT,
    _Float16* __restrict__ h16, int ntasks)
{
  __shared__ _Float16 s_wn[DD * DD];              // 32 KB
  {
    const uint4* g = (const uint4*)wnT;
    uint4* sm = (uint4*)s_wn;
    int t = threadIdx.x;
#pragma unroll
    for (int j = 0; j < 8; ++j) sm[t + 256 * j] = g[t + 256 * j];   // 2048 uint4
  }
  __syncthreads();

  int task = blockIdx.x * (blockDim.x >> 5) + (threadIdx.x >> 5);   // row-tile id
  if (task >= ntasks) return;                     // wave-uniform; EXEC all-ones
  int lane = threadIdx.x & 31;
  int lo = lane & 15, hi = lane >> 4;

  const _Float16* arow = x16 + (size_t)(task * 16 + lo) * DD;
  v8f zero = {};
  v8f accv[8];
#pragma unroll
  for (int c = 0; c < 8; ++c) accv[c] = zero;

#pragma unroll
  for (int si = 0; si < 4; ++si) {
    int s = si * 32;
    v16h a = load_a_tile(arow, s, hi);
#pragma unroll
    for (int c = 0; c < 8; ++c) {
      v16h b = load_b_16k(&s_wn[(c * 16 + lo) * DD + s + 16 * hi]);
      accv[c] = WMMA_F32_F16(a, b, accv[c]);
    }
  }
  // D layout: m = v + 8*hi, n = lo
#pragma unroll
  for (int c = 0; c < 8; ++c)
#pragma unroll
    for (int v = 0; v < 8; ++v) {
      int row = task * 16 + v + 8 * hi;
      h16[(size_t)row * DD + c * 16 + lo] = (_Float16)accv[c][v];
    }
}

// ---------------- edge kernel: one wave per 16-edge block ----------------
// Per-layer weights (We 32KB + W1a/W1b 16KB each = 64KB) staged in LDS and
// shared by the workgroup's 8 waves.

__global__ void __launch_bounds__(256) edge_kernel(
    const _Float16* __restrict__ h16,
    const int* __restrict__ eidx,        // [2,E]: src row then dst row
    const float* __restrict__ ea,        // [E,2]
    const _Float16* __restrict__ w1aT, const _Float16* __restrict__ w1bT,
    const float* __restrict__ b1, const float* __restrict__ w2,
    const float* __restrict__ b2p,
    const _Float16* __restrict__ weT, const float* __restrict__ we2,
    float* __restrict__ acc, int ntasks)
{
  __shared__ _Float16 s_we[DD * DD];    // 32 KB
  __shared__ _Float16 s_w1a[AH * DD];   // 16 KB
  __shared__ _Float16 s_w1b[AH * DD];   // 16 KB
  {
    int t = threadIdx.x;
    const uint4* gw = (const uint4*)weT;
    uint4* sw = (uint4*)s_we;
#pragma unroll
    for (int j = 0; j < 8; ++j) sw[t + 256 * j] = gw[t + 256 * j];  // 2048 uint4
    const uint4* ga = (const uint4*)w1aT;
    uint4* sa = (uint4*)s_w1a;
#pragma unroll
    for (int j = 0; j < 4; ++j) sa[t + 256 * j] = ga[t + 256 * j];  // 1024 uint4
    const uint4* gb = (const uint4*)w1bT;
    uint4* sb = (uint4*)s_w1b;
#pragma unroll
    for (int j = 0; j < 4; ++j) sb[t + 256 * j] = gb[t + 256 * j];
  }
  __syncthreads();

  int task = blockIdx.x * (blockDim.x >> 5) + (threadIdx.x >> 5);
  if (task >= ntasks) return;          // wave-uniform
  int lane = threadIdx.x & 31;
  int lo = lane & 15, hi = lane >> 4;
  int e = task * 16 + lo;

  int sn = eidx[e];                    // source node  -> x_j
  int dn = eidx[EE + e];               // target node  -> x_i
  float ea0 = ea[2 * e];
  float ea1 = ea[2 * e + 1];
  const _Float16* xi_row = h16 + (size_t)dn * DD;
  const _Float16* xj_row = h16 + (size_t)sn * DD;

  v8f zero = {};
  v16h xj_all[4];
  v8f accA[4];
#pragma unroll
  for (int c = 0; c < 4; ++c) accA[c] = zero;

  // a1 = xi @ W1a + xj @ W1b   (16 x 64, f32 acc)
#pragma unroll
  for (int si = 0; si < 4; ++si) {
    int s = si * 32;
    v16h xi = load_a_tile(xi_row, s, hi);
    v16h xj = load_a_tile(xj_row, s, hi);
    xj_all[si] = xj;
#pragma unroll
    for (int c = 0; c < 4; ++c) {
      v16h ba = load_b_16k(&s_w1a[(c * 16 + lo) * DD + s + 16 * hi]);
      accA[c] = WMMA_F32_F16(xi, ba, accA[c]);
      v16h bb = load_b_16k(&s_w1b[(c * 16 + lo) * DD + s + 16 * hi]);
      accA[c] = WMMA_F32_F16(xj, bb, accA[c]);
    }
  }

  // att[m] = sigmoid( sum_n relu(a1[m][n] + b1[n]) * w2[n] + b2 )
  float b2 = b2p[0];
  float att[8];
#pragma unroll
  for (int v = 0; v < 8; ++v) att[v] = 0.f;
#pragma unroll
  for (int c = 0; c < 4; ++c) {
    float w2v = w2[c * 16 + lo];
    float b1v = b1[c * 16 + lo];
#pragma unroll
    for (int v = 0; v < 8; ++v) {
      float t = accA[c][v] + b1v;
      t = t > 0.f ? t : 0.f;
      att[v] += t * w2v;
    }
  }
#pragma unroll
  for (int v = 0; v < 8; ++v) {
    float r = att[v];
    r += __shfl_xor(r, 1, 32);
    r += __shfl_xor(r, 2, 32);
    r += __shfl_xor(r, 4, 32);
    r += __shfl_xor(r, 8, 32);        // sum over n within 16-lane half (rows 8*hi..8*hi+7)
    att[v] = 1.f / (1.f + __expf(-(r + b2)));
  }

  // msg = att * (xj @ We_node + edge_attr @ We_edge)   (16 x 128)
  v8f accM[8];
#pragma unroll
  for (int c = 0; c < 8; ++c) accM[c] = zero;
#pragma unroll
  for (int si = 0; si < 4; ++si) {
    int s = si * 32;
#pragma unroll
    for (int c = 0; c < 8; ++c) {
      v16h bw = load_b_16k(&s_we[(c * 16 + lo) * DD + s + 16 * hi]);
      accM[c] = WMMA_F32_F16(xj_all[si], bw, accM[c]);
    }
  }

  // scatter-add into target rows
#pragma unroll
  for (int v = 0; v < 8; ++v) {
    int m = v + 8 * hi;                 // edge row this lane holds in C/D layout
    float e0m = __shfl(ea0, m, 32);     // sources are lanes 0..15
    float e1m = __shfl(ea1, m, 32);
    int   dm  = __shfl(dn,  m, 32);
    float a   = att[v];
    float* rowp = acc + (size_t)dm * DD;
#pragma unroll
    for (int c = 0; c < 8; ++c) {
      int n = c * 16 + lo;
      float val = a * (accM[c][v] + e0m * we2[n] + e1m * we2[DD + n]);
      __hip_atomic_fetch_add(rowp + n, val, __ATOMIC_RELAXED, __HIP_MEMORY_SCOPE_AGENT);
    }
  }
}

// ---------------- bias + LayerNorm + ReLU + residual ----------------

__global__ void __launch_bounds__(256) finalize_kernel(
    const float* __restrict__ acc, const float* __restrict__ bias,
    const float* __restrict__ gamma, const float* __restrict__ beta,
    const float* __restrict__ xres, float* __restrict__ xout,
    _Float16* __restrict__ x16out, int do_relu, int nrows)
{
  int row = blockIdx.x * (blockDim.x >> 5) + (threadIdx.x >> 5);
  if (row >= nrows) return;
  int lane = threadIdx.x & 31;
  const float* rp = acc + (size_t)row * DD;
  float v[4];
  float s = 0.f, ss = 0.f;
#pragma unroll
  for (int j = 0; j < 4; ++j) {
    int col = lane + j * 32;
    float t = rp[col] + bias[col];
    v[j] = t; s += t; ss += t * t;
  }
#pragma unroll
  for (int off = 16; off >= 1; off >>= 1) {
    s  += __shfl_xor(s,  off, 32);
    ss += __shfl_xor(ss, off, 32);
  }
  float mu  = s * (1.f / DD);
  float var = ss * (1.f / DD) - mu * mu;
  float inv = rsqrtf(var + 1e-5f);
#pragma unroll
  for (int j = 0; j < 4; ++j) {
    int col = lane + j * 32;
    float y = (v[j] - mu) * inv * gamma[col] + beta[col];
    if (do_relu) y = y > 0.f ? y : 0.f;
    y += xres[(size_t)row * DD + col];
    xout[(size_t)row * DD + col] = y;
    if (x16out) x16out[(size_t)row * DD + col] = (_Float16)y;
  }
}

// ---------------- host launch ----------------

extern "C" void kernel_launch(void* const* d_in, const int* in_sizes, int n_in,
                              void* d_out, int out_size, void* d_ws, size_t ws_size,
                              hipStream_t stream)
{
  const float* x     = (const float*)d_in[0];
  const int*   eidx  = (const int*)d_in[1];
  const float* ea    = (const float*)d_in[2];
  const float* wn    = (const float*)d_in[3];
  const float* we    = (const float*)d_in[4];
  const float* w1    = (const float*)d_in[5];
  const float* b1    = (const float*)d_in[6];
  const float* w2    = (const float*)d_in[7];
  const float* b2    = (const float*)d_in[8];
  const float* bias  = (const float*)d_in[9];
  const float* gamma = (const float*)d_in[10];
  const float* beta  = (const float*)d_in[11];
  float* out = (float*)d_out;

  char* ws = (char*)d_ws;
  _Float16* x16  = (_Float16*)(ws + 0);               // 12,800,000 B
  _Float16* h16  = (_Float16*)(ws + 12800000);        // 12,800,000 B
  float*    acc  = (float*)   (ws + 25600000);        // 25,600,000 B
  float*    xcur = (float*)   (ws + 51200000);        // 25,600,000 B
  _Float16* wnT  = (_Float16*)(ws + 76800000);        //     98,304 B
  _Float16* weT  = (_Float16*)(ws + 76898304);        //     98,304 B
  _Float16* w1aT = (_Float16*)(ws + 76996608);        //     49,152 B
  _Float16* w1bT = (_Float16*)(ws + 77045760);        //     49,152 B
  float*    we2  = (float*)   (ws + 77094912);        //      3,072 B

  prep_weights_kernel<<<(LL * DD * DD + 255) / 256, 256, 0, stream>>>(
      wn, we, w1, wnT, weT, we2, w1aT, w1bT);
  f32_to_f16_kernel<<<(NN * DD + 255) / 256, 256, 0, stream>>>(x, x16, NN * DD);

  for (int i = 0; i < LL; ++i) {
    zero_kernel<<<(NN * DD / 4 + 255) / 256, 256, 0, stream>>>((float4*)acc, NN * DD / 4);

    int ntile = NN / 16;                             // 3125 row-block wave tasks
    node_linear_kernel<<<(ntile + 7) / 8, 256, 0, stream>>>(
        x16, wnT + (size_t)i * DD * DD, h16, ntile);

    int etask = EE / 16;                             // 40000 wave tasks
    edge_kernel<<<(etask + 7) / 8, 256, 0, stream>>>(
        h16, eidx, ea,
        w1aT + (size_t)i * AH * DD, w1bT + (size_t)i * AH * DD,
        b1 + i * AH, w2 + i * AH, b2 + i,
        weT + (size_t)i * DD * DD, we2 + i * 2 * DD,
        acc, etask);

    const float* xres = (i == 0) ? x : xcur;
    float* xo = (i == LL - 1) ? out : xcur;
    _Float16* x16o = (i == LL - 1) ? (_Float16*)nullptr : x16;
    finalize_kernel<<<(NN + 7) / 8, 256, 0, stream>>>(
        acc, bias + i * DD, gamma + i * DD, beta + i * DD,
        xres, xo, x16o, (i < LL - 1) ? 1 : 0, NN);
  }
}
